// KGATLayer_25812753449714
// MI455X (gfx1250) — compile-verified, matrix operand order checked by reading, not run on hardware
//
#include <hip/hip_runtime.h>
#include <hip/hip_bf16.h>

typedef __attribute__((ext_vector_type(16))) __bf16 v16bf;
typedef __attribute__((ext_vector_type(8)))  float  v8f;

#define D 128
#define NEG_SLOPE 0.01f

// ---------------- zero workspace ----------------
__global__ void zero_kernel(float4* __restrict__ p, long n4) {
    long i = (long)blockIdx.x * blockDim.x + threadIdx.x;
    long stride = (long)gridDim.x * blockDim.x;
    float4 z = {0.f, 0.f, 0.f, 0.f};
    for (; i < n4; i += stride) p[i] = z;
}

// ---------------- edge-weighted gather-scatter ----------------
// one wave32 per edge; 4 floats/lane = full 128-float row per wave
__global__ void edge_scatter_kernel(const float* __restrict__ x,
                                    const int* __restrict__ src,
                                    const int* __restrict__ dst,
                                    const float* __restrict__ ea,
                                    float* __restrict__ h, int E) {
    long gtid = (long)blockIdx.x * blockDim.x + threadIdx.x;
    int e = (int)(gtid >> 5);
    int lane = (int)(gtid & 31);
    if (e >= E) return;
    int s = src[e];
    int d = dst[e];
    float a = ea[e];
    const float4* xs = (const float4*)(x + (long)s * D);
    float4 v = xs[lane];
    float* hd = h + (long)d * D + lane * 4;
    atomicAdd(hd + 0, v.x * a);
    atomicAdd(hd + 1, v.y * a);
    atomicAdd(hd + 2, v.z * a);
    atomicAdd(hd + 3, v.w * a);
}

// ---------------- fused bi-interaction GEMM epilogue ----------------
// out = LeakyReLU((x+h) @ W1^T + b1) + LeakyReLU((x*h) @ W2^T + b2)
// one wave per 16-node tile; 8 waves (256 thr) per block -> 128 nodes/block
__global__ __launch_bounds__(256) void bi_interact_wmma_kernel(
    const float* __restrict__ x, const float* __restrict__ h,
    const float* __restrict__ W1, const float* __restrict__ b1,
    const float* __restrict__ W2, const float* __restrict__ b2,
    float* __restrict__ out, int N) {
    __shared__ __bf16 w1s[D * D];   // 32 KB bf16
    __shared__ __bf16 w2s[D * D];   // 32 KB bf16

    int tid = threadIdx.x;
    // native f32->bf16 converts (v_cvt_pk_bf16_f32)
    for (int i = tid; i < D * D; i += 256) {
        w1s[i] = (__bf16)W1[i];
        w2s[i] = (__bf16)W2[i];
    }
    __syncthreads();

    int wave  = tid >> 5;
    int lane  = tid & 31;
    int half  = lane >> 4;    // selects K-group per ISA 16-bit A/B layout
    int mlane = lane & 15;    // M (for A) / N (for B) index
    int tile_base = (blockIdx.x * 8 + wave) * 16;
    if (tile_base >= N) return;      // whole-wave uniform exit keeps EXEC all-1s
    bool full = (tile_base + 16 <= N);   // wave-uniform: all 16 rows valid

    union Frag { unsigned int u[8]; v16bf v; };

    // Build A-fragments for u = x+h, v = x*h (4 chunks of K=32)
    Frag aU[4], aV[4];
    int row  = tile_base + mlane;
    int rrow = row < N ? row : (N - 1);   // clamp tail rows; stores are guarded
    const float* xr = x + (long)rrow * D;
    const float* hr = h + (long)rrow * D;
#pragma unroll
    for (int c = 0; c < 4; ++c) {
#pragma unroll
        for (int g = 0; g < 8; ++g) {
            int k = 32 * c + ((g < 4) ? (8 * half + 2 * g)
                                      : (16 + 8 * half + 2 * (g - 4)));
            float2 xv = *(const float2*)(xr + k);
            float2 hv = *(const float2*)(hr + k);
            aU[c].v[2 * g]     = (__bf16)(xv.x + hv.x);
            aU[c].v[2 * g + 1] = (__bf16)(xv.y + hv.y);
            aV[c].v[2 * g]     = (__bf16)(xv.x * hv.x);
            aV[c].v[2 * g + 1] = (__bf16)(xv.y * hv.y);
        }
    }

#pragma unroll
    for (int o = 0; o < 8; ++o) {
        v8f acc1 = {};
        v8f acc2 = {};
#pragma unroll
        for (int c = 0; c < 4; ++c) {
            Frag B1, B2;
#pragma unroll
            for (int g = 0; g < 8; ++g) {
                int k = 32 * c + ((g < 4) ? (8 * half + 2 * g)
                                          : (16 + 8 * half + 2 * (g - 4)));
                int idx = (o * 16 + mlane) * D + k;   // B[k][n] = W[n][k]
                B1.u[g] = *(const unsigned int*)&w1s[idx];
                B2.u[g] = *(const unsigned int*)&w2s[idx];
            }
            acc1 = __builtin_amdgcn_wmma_f32_16x16x32_bf16(
                false, aU[c].v, false, B1.v, (short)0, acc1, false, false);
            acc2 = __builtin_amdgcn_wmma_f32_16x16x32_bf16(
                false, aV[c].v, false, B2.v, (short)0, acc2, false, false);
        }
        int col = o * 16 + mlane;
        float bias1 = b1[col];
        float bias2 = b2[col];
        float* outc = out + (long)(tile_base + 8 * half) * D + col;
        if (full) {
            // fast path (all blocks except the last): no per-element guards
#pragma unroll
            for (int r = 0; r < 8; ++r) {
                float s1 = acc1[r] + bias1;
                s1 = s1 >= 0.f ? s1 : NEG_SLOPE * s1;
                float s2 = acc2[r] + bias2;
                s2 = s2 >= 0.f ? s2 : NEG_SLOPE * s2;
                outc[(long)r * D] = s1 + s2;
            }
        } else {
#pragma unroll
            for (int r = 0; r < 8; ++r) {
                int m = tile_base + r + 8 * half;   // C/D layout: M = r + 8*half
                if (m < N) {
                    float s1 = acc1[r] + bias1;
                    s1 = s1 >= 0.f ? s1 : NEG_SLOPE * s1;
                    float s2 = acc2[r] + bias2;
                    s2 = s2 >= 0.f ? s2 : NEG_SLOPE * s2;
                    out[(long)m * D + col] = s1 + s2;
                }
            }
        }
    }
}

extern "C" void kernel_launch(void* const* d_in, const int* in_sizes, int n_in,
                              void* d_out, int out_size, void* d_ws, size_t ws_size,
                              hipStream_t stream) {
    const float* x  = (const float*)d_in[0];
    const int*   src = (const int*)d_in[1];
    const int*   dst = (const int*)d_in[2];
    const float* ea = (const float*)d_in[3];
    const float* W1 = (const float*)d_in[4];
    const float* b1 = (const float*)d_in[5];
    const float* W2 = (const float*)d_in[6];
    const float* b2 = (const float*)d_in[7];
    float* out = (float*)d_out;

    int N = in_sizes[0] / D;
    int E = in_sizes[1];
    float* h = (float*)d_ws;   // N*D floats = 51.2 MB scratch (fits in 192MB L2)

    long n4 = (long)N * D / 4;
    zero_kernel<<<2048, 256, 0, stream>>>((float4*)h, n4);

    long edge_threads = (long)E * 32;
    int eblocks = (int)((edge_threads + 255) / 256);
    edge_scatter_kernel<<<eblocks, 256, 0, stream>>>(x, src, dst, ea, h, E);

    int nblocks = (N + 127) / 128;
    bi_interact_wmma_kernel<<<nblocks, 256, 0, stream>>>(x, h, W1, b1, W2, b2, out, N);
}
